// CrossAttention_3753801417337
// MI455X (gfx1250) — matrix operand order repaired and checked by reading, not compiled, not run on hardware
//
#include <hip/hip_runtime.h>

// ---------------------------------------------------------------------------
// CrossAttention for MI455X (gfx1250, wave32, WMMA 16x16x32 f16 -> f32 acc)
// B=4, N=2048, C=1024, H=16, D=64
// cvt(fp32->f16) -> GEMM qproj -> GEMM kvproj -> flash attn -> GEMM out+bias
// GEMMs: async double-buffered LDS tiling (ASYNCcnt), 128x128 block tile,
//        32x64 wave tile (8 WMMAs per K-step between waits).
// ---------------------------------------------------------------------------

typedef _Float16 v16h __attribute__((ext_vector_type(16)));
typedef float    v8f  __attribute__((ext_vector_type(8)));
typedef _Float16 h8   __attribute__((ext_vector_type(8)));
typedef float    f8   __attribute__((ext_vector_type(8)));
typedef int      v4i  __attribute__((ext_vector_type(4)));

typedef __attribute__((address_space(1))) v4i* gv4p;   // global 16B chunk
typedef __attribute__((address_space(3))) v4i* lv4p;   // LDS 16B chunk

#define SEQ   2048
#define CH    1024
#define HEADS 16
#define HD    64
#define TPAD  40   // padded LDS tile row: 32 f16 data + 8 pad (80B stride)

static __device__ inline v8f wmma_f16(v16h a, v16h b, v8f c) {
    return __builtin_amdgcn_wmma_f32_16x16x32_f16(
        false, a, false, b, (short)0, c, false, false);
}

// A-fragment (16x32 f16, M x K): lane row = lane&15, khalf = lane>>4.
static __device__ inline v16h load_a_f16(const _Float16* rowp, int khalf) {
    h8 lo = *(const h8*)(rowp + khalf * 16);
    h8 hi = *(const h8*)(rowp + khalf * 16 + 8);
    v16h a;
#pragma unroll
    for (int i = 0; i < 8; ++i) { a[i] = lo[i]; a[i + 8] = hi[i]; }
    return a;
}

// B-fragment (32x16 f16, K x N): lane col = lane&15, elements walk K.
static __device__ inline v16h load_b_f16(const _Float16* colp, int khalf) {
    h8 lo = *(const h8*)(colp + khalf * 16);
    h8 hi = *(const h8*)(colp + khalf * 16 + 8);
    v16h b;
#pragma unroll
    for (int i = 0; i < 8; ++i) { b[i] = lo[i]; b[i + 8] = hi[i]; }
    return b;
}

// ---- gfx1250 async global->LDS copy (ASYNCcnt path) ----
static __device__ inline void async_g2l_b128(const _Float16* gsrc, _Float16* lds) {
#if defined(__has_builtin) && __has_builtin(__builtin_amdgcn_global_load_async_to_lds_b128)
    __builtin_amdgcn_global_load_async_to_lds_b128((gv4p)gsrc, (lv4p)lds, 0, 0);
#else
    unsigned loff = (unsigned)(size_t)lds;
    asm volatile("global_load_async_to_lds_b128 %0, %1, off"
                 :: "v"(loff), "v"(gsrc) : "memory");
#endif
}

static __device__ inline void wait_async0() {
#if defined(__has_builtin) && __has_builtin(__builtin_amdgcn_s_wait_asynccnt)
    __builtin_amdgcn_s_wait_asynccnt(0);
#else
    asm volatile("s_wait_asynccnt 0x0" ::: "memory");
#endif
}

static __device__ inline void wait_ds0() {
    asm volatile("s_wait_dscnt 0x0" ::: "memory");
}

// ---------------------------------------------------------------------------
// Elementwise fp32 -> f16 conversion, vectorized x8.
// ---------------------------------------------------------------------------
__global__ __launch_bounds__(256)
void k_cvt_f32_f16(const float* __restrict__ src, _Float16* __restrict__ dst) {
    size_t i = ((size_t)blockIdx.x * 256 + threadIdx.x) * 8;
    f8 v = *(const f8*)(src + i);
    h8 o;
#pragma unroll
    for (int j = 0; j < 8; ++j) o[j] = (_Float16)v[j];
    *(h8*)(dst + i) = o;
}

// ---------------------------------------------------------------------------
// Stage a 128-row x 32-col f16 tile into LDS via async b128 copies.
// 512 chunks of 16B, 256 threads x 2 chunks.
// ---------------------------------------------------------------------------
static __device__ inline void stage128x32(const _Float16* g, int K, int k0,
                                          _Float16* lds, int tid) {
#pragma unroll
    for (int it = 0; it < 2; ++it) {
        int chunk = tid * 2 + it;          // 0..511
        int r = chunk >> 2, c = chunk & 3; // row, 16B chunk within row
        async_g2l_b128(g + (size_t)r * K + k0 + c * 8, lds + r * TPAD + c * 8);
    }
}

// ---------------------------------------------------------------------------
// Shared GEMM mainloop: 128x128 block tile over K, async double-buffered LDS.
// Wave tile 32M x 64N: 2 A-frags x 4 B-frags -> 8 accumulators.
// Ag/Wg point at the block's first row.
// ---------------------------------------------------------------------------
static __device__ inline void gemm_core(const _Float16* Ag, const _Float16* Wg, int K,
                                        _Float16* Als0, _Float16* Bls0,
                                        _Float16* Als1, _Float16* Bls1,
                                        v8f acc[2][4]) {
    const int tid   = threadIdx.x;
    const int lane  = tid & 31;
    const int wave  = tid >> 5;
    const int l16   = lane & 15;
    const int khalf = lane >> 4;
    const int ma = (wave >> 1) * 32;   // 4 M-waves
    const int na = (wave & 1) * 64;    // 2 N-waves

    _Float16* As = Als0; _Float16* Bs = Bls0;
    _Float16* An = Als1; _Float16* Bn = Bls1;

    // prologue: stage first K-tile
    stage128x32(Ag, K, 0, As, tid);
    stage128x32(Wg, K, 0, Bs, tid);
    wait_async0();
    __syncthreads();

    for (int k0 = 0; k0 < K; k0 += 32) {
        if (k0 + 32 < K) {                       // stage next tile (async)
            stage128x32(Ag, K, k0 + 32, An, tid);
            stage128x32(Wg, K, k0 + 32, Bn, tid);
        }
        // fragments from current LDS buffers
        v16h a0 = load_a_f16(As + (ma + l16) * TPAD, khalf);
        v16h a1 = load_a_f16(As + (ma + 16 + l16) * TPAD, khalf);
        v16h b0 = load_b_f16(Bs + (na + l16) * TPAD, khalf);
        v16h b1 = load_b_f16(Bs + (na + 16 + l16) * TPAD, khalf);
        v16h b2 = load_b_f16(Bs + (na + 32 + l16) * TPAD, khalf);
        v16h b3 = load_b_f16(Bs + (na + 48 + l16) * TPAD, khalf);
        acc[0][0] = wmma_f16(a0, b0, acc[0][0]);
        acc[0][1] = wmma_f16(a0, b1, acc[0][1]);
        acc[0][2] = wmma_f16(a0, b2, acc[0][2]);
        acc[0][3] = wmma_f16(a0, b3, acc[0][3]);
        acc[1][0] = wmma_f16(a1, b0, acc[1][0]);
        acc[1][1] = wmma_f16(a1, b1, acc[1][1]);
        acc[1][2] = wmma_f16(a1, b2, acc[1][2]);
        acc[1][3] = wmma_f16(a1, b3, acc[1][3]);

        wait_ds0();      // my reads of As/Bs complete
        wait_async0();   // my stage of An/Bn landed
        __syncthreads(); // everyone's reads + stages complete
        _Float16* t;
        t = As; As = An; An = t;
        t = Bs; Bs = Bn; Bn = t;
    }
}

// ---------------------------------------------------------------------------
// GEMM: Out[m,n] = sum_k A[m,k] * W[n,k]  (f16 in, f16 out)
// grid = (N/128, M/128), block = 256.
// ---------------------------------------------------------------------------
__global__ __launch_bounds__(256)
void k_gemm_f16(const _Float16* __restrict__ A, const _Float16* __restrict__ W,
                _Float16* __restrict__ Out, int N, int K) {
    __shared__ __align__(16) _Float16 Als[2][128 * TPAD];
    __shared__ __align__(16) _Float16 Bls[2][128 * TPAD];
    const int mblk = blockIdx.y * 128;
    const int nblk = blockIdx.x * 128;

    v8f acc[2][4] = {};
    gemm_core(A + (size_t)mblk * K, W + (size_t)nblk * K, K,
              Als[0], Bls[0], Als[1], Bls[1], acc);

    const int lane  = threadIdx.x & 31;
    const int wave  = threadIdx.x >> 5;
    const int l16   = lane & 15;
    const int khalf = lane >> 4;
    const int ma = (wave >> 1) * 32, na = (wave & 1) * 64;
#pragma unroll
    for (int i = 0; i < 2; ++i)
#pragma unroll
        for (int v = 0; v < 8; ++v) {
            size_t r = (size_t)(mblk + ma + i * 16 + v + 8 * khalf) * N + nblk + na + l16;
            Out[r]      = (_Float16)acc[i][0][v];
            Out[r + 16] = (_Float16)acc[i][1][v];
            Out[r + 32] = (_Float16)acc[i][2][v];
            Out[r + 48] = (_Float16)acc[i][3][v];
        }
}

// Same mainloop, fp32 output + bias (final projection).
__global__ __launch_bounds__(256)
void k_gemm_f16_f32_bias(const _Float16* __restrict__ A, const _Float16* __restrict__ W,
                         const float* __restrict__ bias, float* __restrict__ Out,
                         int N, int K) {
    __shared__ __align__(16) _Float16 Als[2][128 * TPAD];
    __shared__ __align__(16) _Float16 Bls[2][128 * TPAD];
    const int mblk = blockIdx.y * 128;
    const int nblk = blockIdx.x * 128;

    v8f acc[2][4] = {};
    gemm_core(A + (size_t)mblk * K, W + (size_t)nblk * K, K,
              Als[0], Bls[0], Als[1], Bls[1], acc);

    const int lane  = threadIdx.x & 31;
    const int wave  = threadIdx.x >> 5;
    const int l16   = lane & 15;
    const int khalf = lane >> 4;
    const int ma = (wave >> 1) * 32, na = (wave & 1) * 64;
    const float b0 = bias[nblk + na + l16],      b1 = bias[nblk + na + 16 + l16];
    const float b2 = bias[nblk + na + 32 + l16], b3 = bias[nblk + na + 48 + l16];
#pragma unroll
    for (int i = 0; i < 2; ++i)
#pragma unroll
        for (int v = 0; v < 8; ++v) {
            size_t r = (size_t)(mblk + ma + i * 16 + v + 8 * khalf) * N + nblk + na + l16;
            Out[r]      = acc[i][0][v] + b0;
            Out[r + 16] = acc[i][1][v] + b1;
            Out[r + 32] = acc[i][2][v] + b2;
            Out[r + 48] = acc[i][3][v] + b3;
        }
}

// ---------------------------------------------------------------------------
// Flash attention, causal. One wave = 16 query rows; 4 waves/WG = 64 rows.
// grid = (SEQ/64, B*HEADS), block = 128.
// ---------------------------------------------------------------------------
__global__ __launch_bounds__(128)
void k_attn(const _Float16* __restrict__ qlin, const _Float16* __restrict__ kvlin,
            _Float16* __restrict__ y) {
    const int lane  = threadIdx.x & 31;
    const int wave  = threadIdx.x >> 5;   // 0..3
    const int l16   = lane & 15;
    const int khalf = lane >> 4;
    const int b = blockIdx.y >> 4;
    const int h = blockIdx.y & 15;
    const int qr0 = blockIdx.x * 64 + wave * 16;
    const float scale = 0.125f;           // D^-0.5

    const size_t qbase  = (size_t)b * SEQ * CH;
    const size_t kvbase = (size_t)b * SEQ * (2 * CH);

    const _Float16* qrow = qlin + qbase + (size_t)(qr0 + l16) * CH + h * HD;
    v16h qf0 = load_a_f16(qrow, khalf);
    v16h qf1 = load_a_f16(qrow + 32, khalf);

    float mrow[8], lrow[8];
#pragma unroll
    for (int v = 0; v < 8; ++v) { mrow[v] = -1e30f; lrow[v] = 0.0f; }
    v8f acc[4];
#pragma unroll
    for (int t = 0; t < 4; ++t) acc[t] = (v8f){};

    __shared__ __align__(16) _Float16 Pls[4][16 * 32];   // P transpose buffer
    __shared__ __align__(16) _Float16 Vls[4][32 * 64];   // staged V tile

    for (int j0 = 0; j0 < qr0 + 16; j0 += 32) {
        // ---- async-stage V tile [32 keys x 64 d] into LDS (ASYNCcnt path) ----
        wait_ds0();  // previous iteration's LDS gathers must finish first
        {
            const _Float16* vrow =
                kvlin + kvbase + CH + h * HD + (size_t)(j0 + lane) * (2 * CH);
            _Float16* lrowp = &Vls[wave][lane * 64];
#pragma unroll
            for (int c = 0; c < 8; ++c)
                async_g2l_b128(vrow + c * 8, lrowp + c * 8);
        }

        // ---- S = Q K^T for 32 keys, two 16-key subtiles ----
        v8f s[2];
#pragma unroll
        for (int t = 0; t < 2; ++t) {
            const int jj = j0 + t * 16;
            const _Float16* krow =
                kvlin + kvbase + (size_t)(jj + l16) * (2 * CH) + h * HD;
            v8f sv = {};
            sv = wmma_f16(qf0, load_b_f16(krow, khalf), sv);
            sv = wmma_f16(qf1, load_b_f16(krow + 32, khalf), sv);
#pragma unroll
            for (int v = 0; v < 8; ++v) {
                int row = qr0 + v + 8 * khalf;
                int col = jj + l16;
                float xv = sv[v] * scale;
                sv[v] = (col > row) ? -1e30f : xv;
            }
            s[t] = sv;
        }

        // ---- online softmax (row r lives across 16 lanes of one half) ----
        float rescale[8];
#pragma unroll
        for (int v = 0; v < 8; ++v) {
            float t = fmaxf(s[0][v], s[1][v]);
#pragma unroll
            for (int off = 1; off < 16; off <<= 1)
                t = fmaxf(t, __shfl_xor(t, off, 32));
            float nm = fmaxf(mrow[v], t);
            float pa = __expf(s[0][v] - nm);
            float pb = __expf(s[1][v] - nm);
            float rs = pa + pb;
#pragma unroll
            for (int off = 1; off < 16; off <<= 1)
                rs += __shfl_xor(rs, off, 32);
            float rc = __expf(mrow[v] - nm);
            lrow[v] = lrow[v] * rc + rs;
            mrow[v] = nm;
            rescale[v] = rc;
            s[0][v] = pa;
            s[1][v] = pb;
        }
#pragma unroll
        for (int t = 0; t < 4; ++t)
#pragma unroll
            for (int v = 0; v < 8; ++v) acc[t][v] *= rescale[v];

        // ---- transpose P (C-layout -> A-layout) through LDS ----
#pragma unroll
        for (int v = 0; v < 8; ++v) {
            int row = v + 8 * khalf;
            Pls[wave][row * 32 + l16]      = (_Float16)s[0][v];
            Pls[wave][row * 32 + 16 + l16] = (_Float16)s[1][v];
        }
        wait_ds0();
        v16h pf = load_a_f16(&Pls[wave][l16 * 32], khalf);

        // ---- O += P (16x32) x V (32x64): B-frags gathered from staged LDS ----
        wait_async0();   // V tile resident in LDS
#pragma unroll
        for (int dt = 0; dt < 4; ++dt) {
            v16h vb;
#pragma unroll
            for (int e = 0; e < 16; ++e)
                vb[e] = Vls[wave][(khalf * 16 + e) * 64 + dt * 16 + l16];
            acc[dt] = wmma_f16(pf, vb, acc[dt]);
        }
    }

    // ---- epilogue: O / l -> y ----
#pragma unroll
    for (int v = 0; v < 8; ++v) {
        float inv = 1.0f / lrow[v];
        int row = qr0 + v + 8 * khalf;
        _Float16* yp = y + qbase + (size_t)row * CH + h * HD;
#pragma unroll
        for (int dt = 0; dt < 4; ++dt)
            yp[dt * 16 + l16] = (_Float16)(acc[dt][v] * inv);
    }
}

// ---------------------------------------------------------------------------
extern "C" void kernel_launch(void* const* d_in, const int* in_sizes, int n_in,
                              void* d_out, int out_size, void* d_ws, size_t ws_size,
                              hipStream_t stream) {
    (void)in_sizes; (void)n_in; (void)out_size; (void)ws_size;
    const float* q   = (const float*)d_in[0];   // (B,N,C)
    const float* x   = (const float*)d_in[1];   // (B,N,C)
    const float* wq  = (const float*)d_in[2];   // (C,C)
    const float* wkv = (const float*)d_in[3];   // (2C,C)
    const float* pw  = (const float*)d_in[4];   // (C,C)
    const float* pb  = (const float*)d_in[5];   // (C,)
    float* out = (float*)d_out;

    const size_t BN = (size_t)4 * SEQ;          // 8192 rows
    _Float16* qf    = (_Float16*)d_ws;          // BN x CH         (8M)
    _Float16* xf    = qf    + BN * CH;          // BN x CH         (8M)
    _Float16* wqf   = xf    + BN * CH;          // CH x CH         (1M)
    _Float16* wkvf  = wqf   + (size_t)CH * CH;  // 2CH x CH        (2M)
    _Float16* pwf   = wkvf  + (size_t)2 * CH * CH; // CH x CH      (1M)
    _Float16* qlin  = pwf   + (size_t)CH * CH;  // BN x CH         (8M)
    _Float16* kvlin = qlin  + BN * CH;          // BN x 2CH        (16M)
    _Float16* ybuf  = kvlin + BN * 2 * CH;      // BN x CH         (8M)

    const int TPB = 256, VEC = 8;
    k_cvt_f32_f16<<<(BN * CH) / (TPB * VEC), TPB, 0, stream>>>(q, qf);
    k_cvt_f32_f16<<<(BN * CH) / (TPB * VEC), TPB, 0, stream>>>(x, xf);
    k_cvt_f32_f16<<<((size_t)CH * CH) / (TPB * VEC), TPB, 0, stream>>>(wq, wqf);
    k_cvt_f32_f16<<<((size_t)2 * CH * CH) / (TPB * VEC), TPB, 0, stream>>>(wkv, wkvf);
    k_cvt_f32_f16<<<((size_t)CH * CH) / (TPB * VEC), TPB, 0, stream>>>(pw, pwf);

    // Q projection: (8192 x 1024)
    k_gemm_f16<<<dim3(CH / 128, BN / 128), 256, 0, stream>>>(qf, wqf, qlin, CH, CH);
    // KV projection: (8192 x 2048)
    k_gemm_f16<<<dim3(2 * CH / 128, BN / 128), 256, 0, stream>>>(xf, wkvf, kvlin, 2 * CH, CH);
    // Causal flash attention
    k_attn<<<dim3(SEQ / 64, 4 * HEADS), 128, 0, stream>>>(qlin, kvlin, ybuf);
    // Output projection + bias (fp32 out)
    k_gemm_f16_f32_bias<<<dim3(CH / 128, BN / 128), 256, 0, stream>>>(ybuf, pwf, pb, out, CH, CH);
}
